// Seq2Seq_37434934952023
// MI455X (gfx1250) — matrix operand order repaired and checked by reading, not compiled, not run on hardware
//
#include <hip/hip_runtime.h>
#include <hip/hip_bf16.h>

// ---------------- problem constants ----------------
#define B_    1000
#define T_    100
#define I_    264
#define H_    100
#define G_    400      // 4*H
#define O_    264
#define IP    288      // I padded to 9*32 (K for x-GEMM)
#define HP    128      // H padded to 4*32 (K for h-GEMM)
#define KT_I  9        // k-tiles of 32 over IP
#define KT_H  4        // k-tiles of 32 over HP
#define NT_G  25       // n-tiles of 16 over 4H=400
#define NT_O  17       // n-tiles of 16 over O padded 272
#define OP    272
#define ROWS  16       // batch rows per workgroup (one WMMA M tile)
#define THREADS 256
#define WAVES 8
#define NWG   ((B_ + ROWS - 1) / ROWS)   // 63

typedef __attribute__((ext_vector_type(16))) _Float16 v16h;
typedef __attribute__((ext_vector_type(8)))  _Float16 v8h;
typedef __attribute__((ext_vector_type(8)))  float    v8f;

// ---------------- fragment loaders ----------------
// A fragment (16x32 f16) from LDS row-major buffer [16][KP].
// 16-bit A layout: lanes 0-15 hold M=lane, K = {0..7, 16..23};
// lanes 16-31 hold M=lane-16, K = {8..15, 24..31}.
__device__ __forceinline__ v16h loadA(const _Float16* buf, int KP, int kt, int lane) {
    int m  = lane & 15;
    int kh = (lane >> 4) ? 8 : 0;
    const _Float16* p = buf + m * KP + kt * 32 + kh;
    union { v16h v; v8h h[2]; } u;
    u.h[0] = *(const v8h*)(p);        // K = kt*32 + kh + 0..7
    u.h[1] = *(const v8h*)(p + 16);   // K = kt*32 + kh + 16..23
    return u.v;
}

// B fragment (32x16 f16) from packed global weights: tile = 512 halves,
// lane l holds column l%16, K = {(l/16)*8 + 0..7, 16 + (l/16)*8 + 0..7}.
__device__ __forceinline__ v16h loadB(const _Float16* w, int tile, int lane) {
    return *(const v16h*)(w + (size_t)tile * 512 + lane * 16);
}

__device__ __forceinline__ v8f mma(v16h a, v16h b, v8f c) {
    return __builtin_amdgcn_wmma_f32_16x16x32_f16(false, a, false, b, (short)0, c, false, false);
}

__device__ __forceinline__ float sigf(float x) { return 1.0f / (1.0f + __expf(-x)); }

__device__ __forceinline__ void storeD(float* dst, int ldd, int nt, int lane,
                                       const v8f& acc, const float* bias) {
    int col   = nt * 16 + (lane & 15);
    int rbase = (lane < 16) ? 0 : 8;   // D layout: VGPR r -> M = rbase + r
    float b = bias[col];
#pragma unroll
    for (int r = 0; r < 8; ++r)
        dst[(rbase + r) * ldd + col] = acc[r] + b;
}

// ---------------- gates GEMM ----------------
// 25 n-tiles over 8 waves: tiles 3w,3w+1,3w+2 (3 independent WMMA chains,
// shared A-fragments), leftover tile 24 on wave 7 under a *scalar* branch.
__device__ __forceinline__ void lstm_gates(const _Float16* xA, const _Float16* hA,
                                           const _Float16* Wih_p, const _Float16* Whh_p,
                                           const float* bc, float* gates,
                                           int wave, int lane) {
    const int nt0 = wave * 3;
    v8f acc0 = {}, acc1 = {}, acc2 = {};
#pragma unroll
    for (int kt = 0; kt < KT_I; ++kt) {
        v16h a = loadA(xA, IP, kt, lane);
        acc0 = mma(a, loadB(Wih_p, kt * NT_G + nt0,     lane), acc0);
        acc1 = mma(a, loadB(Wih_p, kt * NT_G + nt0 + 1, lane), acc1);
        acc2 = mma(a, loadB(Wih_p, kt * NT_G + nt0 + 2, lane), acc2);
    }
#pragma unroll
    for (int kt = 0; kt < KT_H; ++kt) {
        v16h a = loadA(hA, HP, kt, lane);
        acc0 = mma(a, loadB(Whh_p, kt * NT_G + nt0,     lane), acc0);
        acc1 = mma(a, loadB(Whh_p, kt * NT_G + nt0 + 1, lane), acc1);
        acc2 = mma(a, loadB(Whh_p, kt * NT_G + nt0 + 2, lane), acc2);
    }
    storeD(gates, G_, nt0,     lane, acc0, bc);
    storeD(gates, G_, nt0 + 1, lane, acc1, bc);
    storeD(gates, G_, nt0 + 2, lane, acc2, bc);
    if (wave == 7) {          // scalar branch (wave is readfirstlane'd)
        v8f acc3 = {};
#pragma unroll
        for (int kt = 0; kt < KT_I; ++kt)
            acc3 = mma(loadA(xA, IP, kt, lane), loadB(Wih_p, kt * NT_G + 24, lane), acc3);
#pragma unroll
        for (int kt = 0; kt < KT_H; ++kt)
            acc3 = mma(loadA(hA, HP, kt, lane), loadB(Whh_p, kt * NT_G + 24, lane), acc3);
        storeD(gates, G_, 24, lane, acc3, bc);
    }
}

// ---------------- elementwise LSTM cell update ----------------
__device__ __forceinline__ void lstm_update(const float* gates, float* cB, _Float16* hA, int tid) {
    for (int idx = tid; idx < ROWS * H_; idx += THREADS) {
        int row = idx / H_, j = idx - row * H_;
        float i = sigf(gates[row * G_ + j]);
        float f = sigf(gates[row * G_ + H_ + j]);
        float g = tanhf(gates[row * G_ + 2 * H_ + j]);
        float o = sigf(gates[row * G_ + 3 * H_ + j]);
        float c = f * cB[idx] + i * g;
        float h = o * tanhf(c);
        cB[idx] = c;
        hA[row * HP + j] = (_Float16)h;
    }
}

// ---------------- FC epilogue: bias, next-step input, loss ----------------
__device__ __forceinline__ void fc_epilogue(const v8f& acc, int nt, int lane,
                                            const float* fcb, _Float16* xA,
                                            const float* tgtBase, int validRows,
                                            float* lossLocal) {
    int col   = nt * 16 + (lane & 15);
    int rbase = (lane < 16) ? 0 : 8;
    if (col < O_) {
        float bias = fcb[col];
#pragma unroll
        for (int r = 0; r < 8; ++r) {
            int row = rbase + r;
            float p = acc[r] + bias;
            xA[row * IP + col] = (_Float16)p;   // feed next-step input
            if (row < validRows) {
                float d = p - tgtBase[(size_t)row * (T_ * I_) + col];
                *lossLocal += d * d;
            }
        }
    }
}

// 17 n-tiles over 8 waves: tiles 2w,2w+1; leftover tile 16 on wave 7 (scalar).
__device__ __forceinline__ void fc_step(const _Float16* hA, const _Float16* Fc_p,
                                        const float* fcb, _Float16* xA,
                                        const float* tgtBase, int validRows,
                                        float* lossLocal, int wave, int lane) {
    const int nt0 = wave * 2;
    v8f acc0 = {}, acc1 = {};
#pragma unroll
    for (int kt = 0; kt < KT_H; ++kt) {
        v16h a = loadA(hA, HP, kt, lane);
        acc0 = mma(a, loadB(Fc_p, kt * NT_O + nt0,     lane), acc0);
        acc1 = mma(a, loadB(Fc_p, kt * NT_O + nt0 + 1, lane), acc1);
    }
    fc_epilogue(acc0, nt0,     lane, fcb, xA, tgtBase, validRows, lossLocal);
    fc_epilogue(acc1, nt0 + 1, lane, fcb, xA, tgtBase, validRows, lossLocal);
    if (wave == 7) {          // scalar branch
        v8f acc2 = {};
#pragma unroll
        for (int kt = 0; kt < KT_H; ++kt)
            acc2 = mma(loadA(hA, HP, kt, lane), loadB(Fc_p, kt * NT_O + 16, lane), acc2);
        fc_epilogue(acc2, 16, lane, fcb, xA, tgtBase, validRows, lossLocal);
    }
}

// ---------------- kernel 1: pack weights (f16 WMMA-B tiles), biases, zero loss ----------------
__device__ __forceinline__ void packW(const float* W, int N, int K, int KTt, int NTt,
                                      _Float16* out, int gtid, int nthreads) {
    int total = KTt * NTt * 512;
    for (int idx = gtid; idx < total; idx += nthreads) {
        int tile = idx >> 9;
        int rem  = idx & 511;
        int l    = rem >> 4;       // lane 0..31
        int kk   = rem & 15;
        int kt = tile / NTt, nt = tile - kt * NTt;
        int n = nt * 16 + (l & 15);
        int k = kt * 32 + ((l >> 4) * 8) + (kk & 7) + ((kk >> 3) * 16);
        float v = (n < N && k < K) ? W[(size_t)n * K + k] : 0.0f;
        out[idx] = (_Float16)v;
    }
}

__global__ void seq2seq_prep(
    const float* enc_Wih, const float* enc_Whh, const float* enc_bih, const float* enc_bhh,
    const float* pd_Wih, const float* pd_Whh, const float* pd_bih, const float* pd_bhh,
    const float* pd_fcW, const float* pd_fcb,
    const float* rd_Wih, const float* rd_Whh, const float* rd_bih, const float* rd_bhh,
    const float* rd_fcW, const float* rd_fcb,
    _Float16* encWih_p, _Float16* encWhh_p,
    _Float16* pdWih_p, _Float16* pdWhh_p, _Float16* pdFc_p,
    _Float16* rdWih_p, _Float16* rdWhh_p, _Float16* rdFc_p,
    float* bc_enc, float* bc_pd, float* bc_rd, float* fcb_pd_o, float* fcb_rd_o,
    float* lossAccum) {
    int gtid = blockIdx.x * blockDim.x + threadIdx.x;
    int nthreads = gridDim.x * blockDim.x;
    if (gtid < 2) lossAccum[gtid] = 0.0f;   // re-zeroed every launch (graph replay safe)
    packW(enc_Wih, G_, I_, KT_I, NT_G, encWih_p, gtid, nthreads);
    packW(enc_Whh, G_, H_, KT_H, NT_G, encWhh_p, gtid, nthreads);
    packW(pd_Wih,  G_, I_, KT_I, NT_G, pdWih_p,  gtid, nthreads);
    packW(pd_Whh,  G_, H_, KT_H, NT_G, pdWhh_p,  gtid, nthreads);
    packW(pd_fcW,  O_, H_, KT_H, NT_O, pdFc_p,   gtid, nthreads);
    packW(rd_Wih,  G_, I_, KT_I, NT_G, rdWih_p,  gtid, nthreads);
    packW(rd_Whh,  G_, H_, KT_H, NT_G, rdWhh_p,  gtid, nthreads);
    packW(rd_fcW,  O_, H_, KT_H, NT_O, rdFc_p,   gtid, nthreads);
    for (int j = gtid; j < G_; j += nthreads) {
        bc_enc[j] = enc_bih[j] + enc_bhh[j];
        bc_pd[j]  = pd_bih[j]  + pd_bhh[j];
        bc_rd[j]  = rd_bih[j]  + rd_bhh[j];
    }
    for (int j = gtid; j < OP; j += nthreads) {
        fcb_pd_o[j] = (j < O_) ? pd_fcb[j] : 0.0f;
        fcb_rd_o[j] = (j < O_) ? rd_fcb[j] : 0.0f;
    }
}

// ---------------- kernel 2: encoder (double-buffered async x staging) ----------------
__global__ __launch_bounds__(THREADS) void seq2seq_encoder(
    const float* src,
    const _Float16* encWih_p, const _Float16* encWhh_p, const float* bc_enc,
    _Float16* hEncWS, float* cEncWS) {
    __shared__ alignas(16) _Float16 xA[ROWS * IP];    //  9216 B
    __shared__ alignas(16) _Float16 hA[ROWS * HP];    //  4096 B
    __shared__ float gates[ROWS * G_];                // 25600 B
    __shared__ float cB[ROWS * H_];                   //  6400 B
    __shared__ alignas(16) float stage[ROWS * I_];    // 16896 B  (fp32 x staging)

    const int tid  = threadIdx.x;
    const int lane = tid & 31;
    const int wave = __builtin_amdgcn_readfirstlane(tid >> 5);  // provably scalar
    const int b0   = blockIdx.x * ROWS;
    const int validRows = (B_ - b0 < ROWS) ? (B_ - b0) : ROWS;
    const uint32_t stageLds = (uint32_t)(uintptr_t)&stage[0];

    // init h=0, c=0
    for (int idx = tid; idx < ROWS * HP; idx += THREADS) hA[idx] = (_Float16)0.0f;
    for (int idx = tid; idx < ROWS * H_; idx += THREADS) cB[idx] = 0.0f;

    // async-stage src rows for timestep t into `stage` (CDNA5 async-to-LDS path)
    auto issue_async_x = [&](int t) {
        for (int c = tid; c < ROWS * 66; c += THREADS) {   // 66 x 16B chunks per row
            int row = c / 66, cw = c - row * 66;
            if (row < validRows) {
                uint64_t g = (uint64_t)(uintptr_t)(src +
                             ((size_t)(b0 + row) * T_ + t) * I_ + cw * 4);
                uint32_t l = stageLds + (uint32_t)(row * I_ + cw * 4) * 4u;
                asm volatile("global_load_async_to_lds_b128 %0, %1, off"
                             :: "v"(l), "v"(g) : "memory");
            }
        }
    };

    issue_async_x(0);
    asm volatile("s_wait_asynccnt 0x0" ::: "memory");
    __syncthreads();

    for (int t = 0; t < T_; ++t) {
        // convert staged fp32 -> f16 A-operand layout
        for (int idx = tid; idx < ROWS * IP; idx += THREADS) {
            int row = idx / IP, col = idx - row * IP;
            float v = (col < I_ && row < validRows) ? stage[row * I_ + col] : 0.0f;
            xA[idx] = (_Float16)v;
        }
        __syncthreads();                  // xA ready; stage free for next step
        if (t + 1 < T_) issue_async_x(t + 1);   // overlaps with WMMA below
        lstm_gates(xA, hA, encWih_p, encWhh_p, bc_enc, gates, wave, lane);
        __syncthreads();
        lstm_update(gates, cB, hA, tid);
        asm volatile("s_wait_asynccnt 0x0" ::: "memory");  // next stage complete (this wave)
        __syncthreads();                                    // ... and all waves
    }

    // spill encoder final state for decoder kernel
    for (int idx = tid; idx < ROWS * HP; idx += THREADS)
        hEncWS[(size_t)blockIdx.x * ROWS * HP + idx] = hA[idx];
    for (int idx = tid; idx < ROWS * H_; idx += THREADS)
        cEncWS[(size_t)blockIdx.x * ROWS * H_ + idx] = cB[idx];
}

// ---------------- kernel 3: both decoders concurrently (2x grid) ----------------
__global__ __launch_bounds__(THREADS) void seq2seq_decoder(
    const float* src, const float* trg,
    const _Float16* pdWih_p, const _Float16* pdWhh_p, const float* bc_pd,
    const _Float16* pdFc_p, const float* fcb_pd,
    const _Float16* rdWih_p, const _Float16* rdWhh_p, const float* bc_rd,
    const _Float16* rdFc_p, const float* fcb_rd,
    const _Float16* hEncWS, const float* cEncWS, float* lossAccum) {
    __shared__ alignas(16) _Float16 xA[ROWS * IP];
    __shared__ alignas(16) _Float16 hA[ROWS * HP];
    __shared__ float gates[ROWS * G_];
    __shared__ float cB[ROWS * H_];

    const int tid  = threadIdx.x;
    const int lane = tid & 31;
    const int wave = __builtin_amdgcn_readfirstlane(tid >> 5);  // provably scalar
    const int dec  = (blockIdx.x >= NWG) ? 1 : 0;   // 0 = predict, 1 = reconstruct
    const int wg   = blockIdx.x - dec * NWG;
    const int b0   = wg * ROWS;
    const int validRows = (B_ - b0 < ROWS) ? (B_ - b0) : ROWS;

    const _Float16* Wih_p = dec ? rdWih_p : pdWih_p;
    const _Float16* Whh_p = dec ? rdWhh_p : pdWhh_p;
    const float*    bc    = dec ? bc_rd   : bc_pd;
    const _Float16* Fc_p  = dec ? rdFc_p  : pdFc_p;
    const float*    fcb   = dec ? fcb_rd  : fcb_pd;
    const float*    tgt0  = (dec ? src : trg) + (size_t)b0 * T_ * I_;

    // restore encoder state; x0 = 0
    for (int idx = tid; idx < ROWS * HP; idx += THREADS)
        hA[idx] = hEncWS[(size_t)wg * ROWS * HP + idx];
    for (int idx = tid; idx < ROWS * H_; idx += THREADS)
        cB[idx] = cEncWS[(size_t)wg * ROWS * H_ + idx];
    for (int idx = tid; idx < ROWS * IP; idx += THREADS)
        xA[idx] = (_Float16)0.0f;
    __syncthreads();

    float lossLocal = 0.0f;
    for (int t = 0; t < T_; ++t) {
        int tt = dec ? (T_ - 1 - t) : t;
        if (tid < validRows)   // warm L2 for this step's target rows
            __builtin_prefetch(tgt0 + (size_t)tid * (T_ * I_) + (size_t)tt * I_, 0, 1);
        lstm_gates(xA, hA, Wih_p, Whh_p, bc, gates, wave, lane);
        __syncthreads();
        lstm_update(gates, cB, hA, tid);
        __syncthreads();
        fc_step(hA, Fc_p, fcb, xA, tgt0 + (size_t)tt * I_, validRows,
                &lossLocal, wave, lane);
        __syncthreads();
    }

    // block reduction (reuse gates as scratch)
    gates[tid] = lossLocal;
    __syncthreads();
    for (int s = THREADS / 2; s > 0; s >>= 1) {
        if (tid < s) gates[tid] += gates[tid + s];
        __syncthreads();
    }
    if (tid == 0) atomicAdd(&lossAccum[dec], gates[0]);
}

// ---------------- kernel 4: finalize means ----------------
__global__ void seq2seq_finalize(const float* lossAccum, float* out) {
    if (blockIdx.x == 0 && threadIdx.x == 0) {
        const float denom = (float)B_ * (float)T_ * (float)O_;
        out[0] = lossAccum[1] / denom;   // reconstruct_loss
        out[1] = lossAccum[0] / denom;   // predict_loss
    }
}

// ---------------- host launch ----------------
extern "C" void kernel_launch(void* const* d_in, const int* in_sizes, int n_in,
                              void* d_out, int out_size, void* d_ws, size_t ws_size,
                              hipStream_t stream) {
    const float* src     = (const float*)d_in[0];
    const float* trg     = (const float*)d_in[1];
    const float* enc_Wih = (const float*)d_in[2];
    const float* enc_Whh = (const float*)d_in[3];
    const float* enc_bih = (const float*)d_in[4];
    const float* enc_bhh = (const float*)d_in[5];
    const float* pd_Wih  = (const float*)d_in[6];
    const float* pd_Whh  = (const float*)d_in[7];
    const float* pd_bih  = (const float*)d_in[8];
    const float* pd_bhh  = (const float*)d_in[9];
    const float* pd_fcW  = (const float*)d_in[10];
    const float* pd_fcb  = (const float*)d_in[11];
    const float* rd_Wih  = (const float*)d_in[12];
    const float* rd_Whh  = (const float*)d_in[13];
    const float* rd_bih  = (const float*)d_in[14];
    const float* rd_bhh  = (const float*)d_in[15];
    const float* rd_fcW  = (const float*)d_in[16];
    const float* rd_fcb  = (const float*)d_in[17];

    char* ws = (char*)d_ws;
    float* lossAccum = (float*)ws;
    size_t off = 256;
    const size_t szWih = (size_t)KT_I * NT_G * 512 * sizeof(_Float16);  // 230400
    const size_t szWhh = (size_t)KT_H * NT_G * 512 * sizeof(_Float16);  // 102400
    const size_t szFc  = (size_t)KT_H * NT_O * 512 * sizeof(_Float16);  //  69632
    _Float16* encWih_p = (_Float16*)(ws + off); off += szWih;
    _Float16* encWhh_p = (_Float16*)(ws + off); off += szWhh;
    _Float16* pdWih_p  = (_Float16*)(ws + off); off += szWih;
    _Float16* pdWhh_p  = (_Float16*)(ws + off); off += szWhh;
    _Float16* pdFc_p   = (_Float16*)(ws + off); off += szFc;
    _Float16* rdWih_p  = (_Float16*)(ws + off); off += szWih;
    _Float16* rdWhh_p  = (_Float16*)(ws + off); off += szWhh;
    _Float16* rdFc_p   = (_Float16*)(ws + off); off += szFc;
    float* bc_enc = (float*)(ws + off); off += G_ * sizeof(float);
    float* bc_pd  = (float*)(ws + off); off += G_ * sizeof(float);
    float* bc_rd  = (float*)(ws + off); off += G_ * sizeof(float);
    float* fcb_pd = (float*)(ws + off); off += OP * sizeof(float);
    float* fcb_rd = (float*)(ws + off); off += OP * sizeof(float);
    off = (off + 255) & ~(size_t)255;
    _Float16* hEncWS = (_Float16*)(ws + off); off += (size_t)NWG * ROWS * HP * sizeof(_Float16);
    float*    cEncWS = (float*)(ws + off);    off += (size_t)NWG * ROWS * H_ * sizeof(float);

    seq2seq_prep<<<64, THREADS, 0, stream>>>(
        enc_Wih, enc_Whh, enc_bih, enc_bhh,
        pd_Wih, pd_Whh, pd_bih, pd_bhh, pd_fcW, pd_fcb,
        rd_Wih, rd_Whh, rd_bih, rd_bhh, rd_fcW, rd_fcb,
        encWih_p, encWhh_p, pdWih_p, pdWhh_p, pdFc_p,
        rdWih_p, rdWhh_p, rdFc_p,
        bc_enc, bc_pd, bc_rd, fcb_pd, fcb_rd, lossAccum);

    seq2seq_encoder<<<NWG, THREADS, 0, stream>>>(
        src, encWih_p, encWhh_p, bc_enc, hEncWS, cEncWS);

    seq2seq_decoder<<<2 * NWG, THREADS, 0, stream>>>(
        src, trg,
        pdWih_p, pdWhh_p, bc_pd, pdFc_p, fcb_pd,
        rdWih_p, rdWhh_p, bc_rd, rdFc_p, fcb_rd,
        hEncWS, cEncWS, lossAccum);

    seq2seq_finalize<<<1, 64, 0, stream>>>(lossAccum, (float*)d_out);
}
